// CausalSelfAttention_34626026341085
// MI455X (gfx1250) — matrix operand order repaired and checked by reading, not compiled, not run on hardware
//
#include <hip/hip_runtime.h>
#include <hip/hip_bf16.h>

// ---------------------------------------------------------------------------
// CausalSelfAttention for MI455X (gfx1250, wave32, WMMA bf16 path)
// B=2, S=2048, D=1024, H=16, DK=64
// ---------------------------------------------------------------------------

#define B_ 2
#define S_ 2048
#define D_ 1024
#define H_ 16
#define DK_ 64

typedef __attribute__((ext_vector_type(16))) __bf16        v16bf;
typedef __attribute__((ext_vector_type(8)))  float         v8f;
typedef __attribute__((ext_vector_type(8)))  unsigned int  v8u;
typedef __attribute__((ext_vector_type(4)))  unsigned int  v4u;

__device__ __forceinline__ unsigned short f2bf(float f) {
    unsigned int u = __float_as_uint(f);
    u = (u + 0x7FFFu + ((u >> 16) & 1u)) >> 16;   // round-to-nearest-even
    return (unsigned short)u;
}

// Build a 16xbf16 fragment from two 16-byte chunks.
__device__ __forceinline__ v16bf frag_from2(const unsigned short* p0,
                                            const unsigned short* p1) {
    v4u a = *(const v4u*)p0;
    v4u b = *(const v4u*)p1;
    v8u u;
    u[0] = a[0]; u[1] = a[1]; u[2] = a[2]; u[3] = a[3];
    u[4] = b[0]; u[5] = b[1]; u[6] = b[2]; u[7] = b[3];
    return __builtin_bit_cast(v16bf, u);
}

// A-matrix fragment (16x32, M x K): lane = row m (0..15), half = lane/16.
// Lane holds k in [8h,8h+8) and [16+8h, 16+8h+8).  row = ptr to 32-elem K slice.
__device__ __forceinline__ v16bf load_a(const unsigned short* row, int half) {
    return frag_from2(row + 8 * half, row + 16 + 8 * half);
}

// B-matrix fragment (32x16, K x N): lane = column n (0..15), half = lane/16.
// Lane holds k in [8h,8h+8) and [16+8h, 16+8h+8) (same order as load_a).
__device__ __forceinline__ v16bf load_b(const unsigned short* row, int half) {
    return frag_from2(row + 8 * half, row + 16 + 8 * half);
}

__device__ __forceinline__ v8f wmma_bf16(v16bf a, v16bf b, v8f c) {
    return __builtin_amdgcn_wmma_f32_16x16x32_bf16(
        /*neg_a=*/false, a, /*neg_b=*/false, b,
        /*c_mod=*/(short)0, c, /*reuse_a=*/false, /*reuse_b=*/false);
}

// ---------------------------------------------------------------------------
// Conversion kernels
// ---------------------------------------------------------------------------
__global__ __launch_bounds__(256) void conv_plain(const float* __restrict__ in,
                                                  unsigned short* __restrict__ out,
                                                  int n) {
    int i = blockIdx.x * 256 + threadIdx.x;
    if (i < n) out[i] = f2bf(in[i]);
}

// (H, D, DK) -> (H, DK, D) with bf16 conversion
__global__ __launch_bounds__(256) void conv_w_t(const float* __restrict__ in,
                                                unsigned short* __restrict__ out) {
    int i = blockIdx.x * 256 + threadIdx.x;
    const int total = H_ * D_ * DK_;
    if (i >= total) return;
    int h = i / (D_ * DK_);
    int r = i % (D_ * DK_);
    int n = r / D_;
    int d = r % D_;
    out[i] = f2bf(in[((size_t)h * D_ + d) * DK_ + n]);
}

// ---------------------------------------------------------------------------
// QKV projection:  out(b,h)[s, n] = sum_d xb[b,s,d] * wT[h,n,d]
// One wave computes a 32x64 tile (8 WMMAs / K-step, B frags reused 2x).
// Register double-buffered K loop.  grid = (S/256, B*H), block = 256.
// transpose_out: 0 -> (bh, S, DK); 1 -> (bh, DK, S)  (used for V)
// ---------------------------------------------------------------------------
__global__ __launch_bounds__(256) void proj_kernel(
    const unsigned short* __restrict__ xb,   // (B, S, D) bf16
    const unsigned short* __restrict__ wT,   // (H, DK, D) bf16
    unsigned short* __restrict__ out,
    float scale, int transpose_out) {
    int bh   = blockIdx.y;
    int b    = bh >> 4;
    int h    = bh & 15;
    int wave = threadIdx.x >> 5;
    int lane = threadIdx.x & 31;
    int half = lane >> 4;
    int ln   = lane & 15;

    int m0 = blockIdx.x * 256 + wave * 32;
    const unsigned short* arow0 = xb + ((size_t)(b * S_ + m0 + ln)) * D_;
    const unsigned short* arow1 = arow0 + (size_t)16 * D_;
    const unsigned short* brow[4];
#pragma unroll
    for (int t = 0; t < 4; t++)
        brow[t] = wT + ((size_t)h * DK_ + t * 16 + ln) * D_;

    v8f acc[8];
#pragma unroll
    for (int i = 0; i < 8; i++) acc[i] = (v8f)0.f;

    // prologue: fragments for kk = 0
    v16bf a0 = load_a(arow0, half);
    v16bf a1 = load_a(arow1, half);
    v16bf bb[4];
#pragma unroll
    for (int t = 0; t < 4; t++) bb[t] = load_b(brow[t], half);

    for (int kk = 0; kk < D_; kk += 32) {
        int kn = (kk + 32) & (D_ - 1);           // wraps to 0 on last iter
        // prefetch next iteration's fragments (in-bounds, discarded at tail)
        v16bf a0n = load_a(arow0 + kn, half);
        v16bf a1n = load_a(arow1 + kn, half);
        v16bf bbn[4];
#pragma unroll
        for (int t = 0; t < 4; t++) bbn[t] = load_b(brow[t] + kn, half);

#pragma unroll
        for (int t = 0; t < 4; t++) acc[t]     = wmma_bf16(a0, bb[t], acc[t]);
#pragma unroll
        for (int t = 0; t < 4; t++) acc[4 + t] = wmma_bf16(a1, bb[t], acc[4 + t]);

        a0 = a0n; a1 = a1n;
#pragma unroll
        for (int t = 0; t < 4; t++) bb[t] = bbn[t];
    }

    size_t base = (size_t)bh * S_ * DK_;
#pragma unroll
    for (int r = 0; r < 2; r++) {
#pragma unroll
        for (int t = 0; t < 4; t++) {
            int n = t * 16 + ln;
#pragma unroll
            for (int v = 0; v < 8; v++) {
                int srow = m0 + r * 16 + v + 8 * half;
                unsigned short val = f2bf(acc[r * 4 + t][v] * scale);
                if (transpose_out)
                    out[base + (size_t)n * S_ + srow] = val;   // (bh, DK, S)
                else
                    out[base + (size_t)srow * DK_ + n] = val;  // (bh, S, DK)
            }
        }
    }
}

// ---------------------------------------------------------------------------
// Flash attention.  One wave handles 16 query rows, streams 32 keys / step.
// V frags issued early; next K frags prefetched before the softmax VALU block
// so exp/shuffle work hides memory latency.
// grid = B*H*(S/16)/8 = 512 blocks of 256 threads (8 waves).
// ---------------------------------------------------------------------------
__global__ __launch_bounds__(256) void flash_kernel(
    const unsigned short* __restrict__ qb,   // (bh, S, DK), pre-scaled
    const unsigned short* __restrict__ kb,   // (bh, S, DK)
    const unsigned short* __restrict__ vt,   // (bh, DK, S)
    unsigned short* __restrict__ cb) {       // (B, S, H*DK)
    __shared__ unsigned short ldsP[8][16 * 32];   // per-wave P tile (16x32 bf16)

    int wave = threadIdx.x >> 5;
    int lane = threadIdx.x & 31;
    int half = lane >> 4;
    int ln   = lane & 15;

    int gw    = blockIdx.x * 8 + wave;
    int qt    = gw & (S_ / 16 - 1);
    int bh    = gw >> 7;                 // / (S_/16)
    int qbase = qt * 16;
    const int jend = qbase + 16;

    const unsigned short* qrow = qb + ((size_t)bh * S_ + qbase + ln) * DK_;
    v16bf qf0 = load_a(qrow, half);
    v16bf qf1 = load_a(qrow + 32, half);

    const unsigned short* kbh = kb + (size_t)bh * S_ * DK_;
    const unsigned short* vbh = vt + (size_t)bh * DK_ * S_;
    unsigned short* P = &ldsP[wave][0];

    v8f o[4];
#pragma unroll
    for (int t = 0; t < 4; t++) o[t] = (v8f)0.f;
    float mst[8], lst[8];
#pragma unroll
    for (int v = 0; v < 8; v++) { mst[v] = -1e30f; lst[v] = 0.f; }

    // prologue: K fragments for jb = 0  (kf: [tile0 lo, tile0 hi, tile1 lo, tile1 hi])
    v16bf kf[4];
    {
        const unsigned short* kr0 = kbh + (size_t)ln * DK_;
        const unsigned short* kr1 = kbh + (size_t)(16 + ln) * DK_;
        kf[0] = load_b(kr0, half);      kf[1] = load_b(kr0 + 32, half);
        kf[2] = load_b(kr1, half);      kf[3] = load_b(kr1 + 32, half);
    }

    for (int jb = 0; jb < jend; jb += 32) {
        // ---- issue V fragments for this step early (consumed after softmax)
        v16bf vf[4];
#pragma unroll
        for (int t = 0; t < 4; t++)
            vf[t] = load_b(vbh + (size_t)(t * 16 + ln) * S_ + jb, half);

        // ---- scores: S = Q @ K^T for 32 keys (two 16x16 tiles) ----
        v8f s0 = (v8f)0.f, s1 = (v8f)0.f;
        s0 = wmma_bf16(qf0, kf[0], s0);
        s0 = wmma_bf16(qf1, kf[1], s0);
        s1 = wmma_bf16(qf0, kf[2], s1);
        s1 = wmma_bf16(qf1, kf[3], s1);

        // ---- prefetch next step's K fragments (clamped at tail) ----
        int jn = (jb + 32 < jend) ? jb + 32 : jb;
        {
            const unsigned short* kr0 = kbh + (size_t)(jn + ln) * DK_;
            const unsigned short* kr1 = kbh + (size_t)(jn + 16 + ln) * DK_;
            kf[0] = load_b(kr0, half);  kf[1] = load_b(kr0 + 32, half);
            kf[2] = load_b(kr1, half);  kf[3] = load_b(kr1 + 32, half);
        }

        // ---- online softmax (row r = v + 8*half lives in element v) ----
#pragma unroll
        for (int v = 0; v < 8; v++) {
            int q = qbase + v + 8 * half;
            float e0 = (jb + ln      > q) ? -1e30f : s0[v];
            float e1 = (jb + 16 + ln > q) ? -1e30f : s1[v];
            float mx = fmaxf(e0, e1);
            mx = fmaxf(mx, __shfl_xor(mx, 1, 32));
            mx = fmaxf(mx, __shfl_xor(mx, 2, 32));
            mx = fmaxf(mx, __shfl_xor(mx, 4, 32));
            mx = fmaxf(mx, __shfl_xor(mx, 8, 32));
            float mnew  = fmaxf(mst[v], mx);
            float alpha = __expf(mst[v] - mnew);
            float p0 = __expf(e0 - mnew);
            float p1 = __expf(e1 - mnew);
            float rs = p0 + p1;
            rs += __shfl_xor(rs, 1, 32);
            rs += __shfl_xor(rs, 2, 32);
            rs += __shfl_xor(rs, 4, 32);
            rs += __shfl_xor(rs, 8, 32);
            lst[v] = lst[v] * alpha + rs;
            mst[v] = mnew;
            o[0][v] *= alpha; o[1][v] *= alpha; o[2][v] *= alpha; o[3][v] *= alpha;
            int r = v + 8 * half;
            P[r * 32 + ln]      = f2bf(p0);
            P[r * 32 + 16 + ln] = f2bf(p1);
        }

        // ---- O += P @ V (P read back from LDS in A-fragment layout) ----
        v16bf pa = load_a(P + ln * 32, half);
#pragma unroll
        for (int t = 0; t < 4; t++)
            o[t] = wmma_bf16(pa, vf[t], o[t]);
    }

    // ---- epilogue: O / l, store concat layout (B, S, H*DK) ----
    int b = bh >> 4, h = bh & 15;
#pragma unroll
    for (int v = 0; v < 8; v++) {
        float inv = 1.f / lst[v];
        int srow = qbase + v + 8 * half;
        size_t rbase = ((size_t)b * S_ + srow) * D_ + h * DK_;
#pragma unroll
        for (int t = 0; t < 4; t++)
            cb[rbase + t * 16 + ln] = f2bf(o[t][v] * inv);
    }
}

// ---------------------------------------------------------------------------
// Output projection: y[s, n] = sum_d cb[s, d] * wob[n, d]   (y = C @ Wo^T)
// One wave per 32x64 tile, register double-buffered.
// grid = (B*S/256, D/64), block = 256.
// ---------------------------------------------------------------------------
__global__ __launch_bounds__(256) void outproj_kernel(
    const unsigned short* __restrict__ cb,    // (B*S, D) bf16
    const unsigned short* __restrict__ wob,   // (D, D) bf16 row-major
    float* __restrict__ y) {
    int wave = threadIdx.x >> 5;
    int lane = threadIdx.x & 31;
    int half = lane >> 4;
    int ln   = lane & 15;

    int m0 = blockIdx.x * 256 + wave * 32;
    int nb = blockIdx.y * 64;

    const unsigned short* arow0 = cb + (size_t)(m0 + ln) * D_;
    const unsigned short* arow1 = arow0 + (size_t)16 * D_;
    const unsigned short* brow[4];
#pragma unroll
    for (int t = 0; t < 4; t++)
        brow[t] = wob + (size_t)(nb + t * 16 + ln) * D_;

    v8f acc[8];
#pragma unroll
    for (int i = 0; i < 8; i++) acc[i] = (v8f)0.f;

    v16bf a0 = load_a(arow0, half);
    v16bf a1 = load_a(arow1, half);
    v16bf bb[4];
#pragma unroll
    for (int t = 0; t < 4; t++) bb[t] = load_b(brow[t], half);

    for (int kk = 0; kk < D_; kk += 32) {
        int kn = (kk + 32) & (D_ - 1);
        v16bf a0n = load_a(arow0 + kn, half);
        v16bf a1n = load_a(arow1 + kn, half);
        v16bf bbn[4];
#pragma unroll
        for (int t = 0; t < 4; t++) bbn[t] = load_b(brow[t] + kn, half);

#pragma unroll
        for (int t = 0; t < 4; t++) acc[t]     = wmma_bf16(a0, bb[t], acc[t]);
#pragma unroll
        for (int t = 0; t < 4; t++) acc[4 + t] = wmma_bf16(a1, bb[t], acc[4 + t]);

        a0 = a0n; a1 = a1n;
#pragma unroll
        for (int t = 0; t < 4; t++) bb[t] = bbn[t];
    }

#pragma unroll
    for (int r = 0; r < 2; r++) {
#pragma unroll
        for (int t = 0; t < 4; t++) {
            int n = nb + t * 16 + ln;
#pragma unroll
            for (int v = 0; v < 8; v++) {
                int srow = m0 + r * 16 + v + 8 * half;
                y[(size_t)srow * D_ + n] = acc[r * 4 + t][v];
            }
        }
    }
}

// ---------------------------------------------------------------------------
// Host-side launcher
// ---------------------------------------------------------------------------
extern "C" void kernel_launch(void* const* d_in, const int* in_sizes, int n_in,
                              void* d_out, int out_size, void* d_ws, size_t ws_size,
                              hipStream_t stream) {
    (void)in_sizes; (void)n_in; (void)out_size; (void)ws_size;
    const float* x  = (const float*)d_in[0];
    const float* Wq = (const float*)d_in[1];
    const float* Wk = (const float*)d_in[2];
    const float* Wv = (const float*)d_in[3];
    const float* Wo = (const float*)d_in[4];

    unsigned short* ws = (unsigned short*)d_ws;
    const size_t BSD  = (size_t)B_ * S_ * D_;         // 4,194,304
    const size_t HDDK = (size_t)H_ * D_ * DK_;        // 1,048,576
    const size_t BHSD = (size_t)B_ * H_ * S_ * DK_;   // 4,194,304

    unsigned short* xb  = ws;                    // (B, S, D)
    unsigned short* wqT = xb  + BSD;             // (H, DK, D)
    unsigned short* wkT = wqT + HDDK;
    unsigned short* wvT = wkT + HDDK;
    unsigned short* wob = wvT + HDDK;            // (D, D)
    unsigned short* qbf = wob + (size_t)D_ * D_; // (bh, S, DK)
    unsigned short* kbf = qbf + BHSD;            // (bh, S, DK)
    unsigned short* vtf = kbf + BHSD;            // (bh, DK, S)
    unsigned short* cbf = vtf + BHSD;            // (B, S, H*DK)

    conv_plain<<<(int)((BSD + 255) / 256), 256, 0, stream>>>(x, xb, (int)BSD);
    conv_w_t  <<<(int)((HDDK + 255) / 256), 256, 0, stream>>>(Wq, wqT);
    conv_w_t  <<<(int)((HDDK + 255) / 256), 256, 0, stream>>>(Wk, wkT);
    conv_w_t  <<<(int)((HDDK + 255) / 256), 256, 0, stream>>>(Wv, wvT);
    conv_plain<<<(int)((D_ * D_ + 255) / 256), 256, 0, stream>>>(Wo, wob, D_ * D_);

    dim3 pg(S_ / 256, B_ * H_);
    const float qscale = 0.125f;   // 1/sqrt(64)
    proj_kernel<<<pg, 256, 0, stream>>>(xb, wqT, qbf, qscale, 0);
    proj_kernel<<<pg, 256, 0, stream>>>(xb, wkT, kbf, 1.0f, 0);
    proj_kernel<<<pg, 256, 0, stream>>>(xb, wvT, vtf, 1.0f, 1);

    flash_kernel<<<B_ * H_ * (S_ / 16) / 8, 256, 0, stream>>>(qbf, kbf, vtf, cbf);

    dim3 og(B_ * S_ / 256, D_ / 64);
    outproj_kernel<<<og, 256, 0, stream>>>(cbf, wob, (float*)d_out);
}
